// BaichuanAttention_16793322127630
// MI455X (gfx1250) — matrix operand order repaired and checked by reading, not compiled, not run on hardware
//
#include <hip/hip_runtime.h>
#include <hip/hip_bf16.h>

typedef __bf16 v16bf __attribute__((ext_vector_type(16)));
typedef __bf16 v4bf  __attribute__((ext_vector_type(4)));
typedef float  v8f   __attribute__((ext_vector_type(8)));

union BF16Frag { v16bf v; uint4 q[2]; };

__device__ __forceinline__ v8f wmma_bf16(v16bf a, v16bf b, v8f c) {
    return __builtin_amdgcn_wmma_f32_16x16x32_bf16(false, a, false, b, (short)0, c, false, false);
}

__device__ __forceinline__ v8f zero8() {
    v8f z;
#pragma unroll
    for (int i = 0; i < 8; ++i) z[i] = 0.f;
    return z;
}

// ---------------- fp32 -> bf16 conversion ----------------
__global__ void cvt_f32_bf16(const float* __restrict__ s, __bf16* __restrict__ d, long long n4) {
    long long i = (long long)blockIdx.x * blockDim.x + threadIdx.x;
    if (i >= n4) return;
    float4 f = ((const float4*)s)[i];
    v4bf o; o[0] = (__bf16)f.x; o[1] = (__bf16)f.y; o[2] = (__bf16)f.z; o[3] = (__bf16)f.w;
    ((v4bf*)d)[i] = o;
}

// ---------------- GEMM: C[M,N] = A[M,K] * W[N,K]^T (torch linear) ----------------
// 256 threads = 8 waves; tile 128x128x32; each wave -> 32x64 (2x4 C frags)
template<bool OUT_F32>
__global__ __launch_bounds__(256) void gemm_xwt(const __bf16* __restrict__ A,
                                                const __bf16* __restrict__ W,
                                                void* __restrict__ Cout,
                                                int M, int N, int K) {
    constexpr int LDT = 40;                         // 32 + 8 pad (keeps 16B alignment)
    __shared__ __align__(16) __bf16 As[128 * LDT];
    __shared__ __align__(16) __bf16 Ws[128 * LDT];
    const int tid  = threadIdx.x;
    const int lane = tid & 31;
    const int wave = tid >> 5;
    const int wm   = wave & 3;                      // 4 M sub-blocks of 32 rows
    const int wn   = wave >> 2;                     // 2 N sub-blocks of 64 cols
    const int l16  = lane & 15;
    const int hs   = lane >> 4;
    const int rowBase = blockIdx.y * 128;
    const int colBase = blockIdx.x * 128;

    v8f acc[2][4];
#pragma unroll
    for (int i = 0; i < 2; ++i)
#pragma unroll
        for (int j = 0; j < 4; ++j) acc[i][j] = zero8();

    for (int k0 = 0; k0 < K; k0 += 32) {
#pragma unroll
        for (int i = 0; i < 2; ++i) {               // 512 8-elem chunks over 256 thr
            int c  = tid + i * 256;
            int r  = c >> 2;
            int c8 = (c & 3) * 8;
            *(uint4*)(As + r * LDT + c8) =
                *(const uint4*)(A + (long long)(rowBase + r) * K + k0 + c8);
            *(uint4*)(Ws + r * LDT + c8) =
                *(const uint4*)(W + (long long)(colBase + r) * K + k0 + c8);
        }
        __syncthreads();
        BF16Frag af[2];
#pragma unroll
        for (int fm = 0; fm < 2; ++fm) {            // A frag: K chunks per ISA layout
            int r = wm * 32 + fm * 16 + l16;
            af[fm].q[0] = *(const uint4*)(As + r * LDT + hs * 8);
            af[fm].q[1] = *(const uint4*)(As + r * LDT + hs * 8 + 16);
        }
        BF16Frag bf[4];
#pragma unroll
        for (int fn = 0; fn < 4; ++fn) {            // B frag: lane%16 = column (=W row)
            int r = wn * 64 + fn * 16 + l16;
            bf[fn].q[0] = *(const uint4*)(Ws + r * LDT + hs * 8);
            bf[fn].q[1] = *(const uint4*)(Ws + r * LDT + hs * 8 + 16);
        }
#pragma unroll
        for (int fm = 0; fm < 2; ++fm)
#pragma unroll
            for (int fn = 0; fn < 4; ++fn)
                acc[fm][fn] = wmma_bf16(af[fm].v, bf[fn].v, acc[fm][fn]);
        __syncthreads();
    }
#pragma unroll
    for (int fm = 0; fm < 2; ++fm) {
        int m0 = rowBase + wm * 32 + fm * 16 + hs * 8;
#pragma unroll
        for (int fn = 0; fn < 4; ++fn) {
            int col = colBase + wn * 64 + fn * 16 + l16;
#pragma unroll
            for (int v = 0; v < 8; ++v) {
                long long idx = (long long)(m0 + v) * N + col;
                if (OUT_F32) ((float*)Cout)[idx] = acc[fm][fn][v];
                else         ((__bf16*)Cout)[idx] = (__bf16)acc[fm][fn][v];
            }
        }
    }
}

// ---------------- Flash attention with analytic ALiBi ----------------
// block = 128 thr (4 waves); block handles 64 Q rows of one (b,h); waves own 16 rows
__global__ __launch_bounds__(128) void flash_attn(const __bf16* __restrict__ qkv,
                                                  __bf16* __restrict__ attn) {
    constexpr int S = 1024, H3 = 15360, HD = 128;
    const int qb = blockIdx.x, h = blockIdx.y, b = blockIdx.z;
    const int tid = threadIdx.x;
    const int lane = tid & 31, wave = tid >> 5;
    const int l16 = lane & 15, hs = lane >> 4;

    __shared__ __align__(16) __bf16 Ks[32 * 136];   // K tile row-major [key][dim]
    __shared__ __align__(16) __bf16 Vt[128 * 40];   // V tile transposed [dim][key]
    __shared__ __align__(16) __bf16 Ps[4][16 * 40]; // per-wave P staging

    const long long rowG = (long long)b * S;
    const int qoff = h * HD, koff = 5120 + h * HD, voff = 10240 + h * HD;

    float slope = (h < 32) ? exp2f(-0.25f * (float)(h + 1))
                           : exp2f(-(2.0f * (float)(h - 32) + 1.0f) * 0.125f);
    const float scale = 0.08838834764831845f;       // 1/sqrt(128)

    // Q fragments: 4 head-dim chunks of 32, kept in registers for all K blocks
    BF16Frag qf[4];
    {
        long long qr = rowG + qb * 64 + wave * 16 + l16;
        const __bf16* qp = qkv + qr * H3 + qoff;
#pragma unroll
        for (int c = 0; c < 4; ++c) {
            qf[c].q[0] = *(const uint4*)(qp + c * 32 + hs * 8);
            qf[c].q[1] = *(const uint4*)(qp + c * 32 + hs * 8 + 16);
        }
    }

    float Mrow[8], Lrow[8];
#pragma unroll
    for (int v = 0; v < 8; ++v) { Mrow[v] = -3.0e38f; Lrow[v] = 0.f; }
    v8f O[8];
#pragma unroll
    for (int f = 0; f < 8; ++f) O[f] = zero8();

    const int nkb = 2 * qb + 2;                     // causal: skip fully-masked blocks
    for (int kb = 0; kb < nkb; ++kb) {
        const int key0 = kb * 32;
#pragma unroll
        for (int i = 0; i < 4; ++i) {               // stage K and V^T tiles
            int c = tid + i * 128;
            int r = c >> 4;                         // key 0..31
            int c8 = (c & 15) * 8;                  // dim chunk
            long long gr = rowG + key0 + r;
            *(uint4*)(Ks + r * 136 + c8) = *(const uint4*)(qkv + gr * H3 + koff + c8);
            union { uint4 u; __bf16 e[8]; } vu;
            vu.u = *(const uint4*)(qkv + gr * H3 + voff + c8);
#pragma unroll
            for (int j = 0; j < 8; ++j) Vt[(c8 + j) * 40 + r] = vu.e[j];
        }
        __syncthreads();

        v8f sfr[2];
#pragma unroll
        for (int f = 0; f < 2; ++f) {               // scores: contract over head dim
            v8f s = zero8();
#pragma unroll
            for (int c = 0; c < 4; ++c) {
                BF16Frag kf;
                int r = f * 16 + l16;
                kf.q[0] = *(const uint4*)(Ks + r * 136 + c * 32 + hs * 8);
                kf.q[1] = *(const uint4*)(Ks + r * 136 + c * 32 + hs * 8 + 16);
                s = wmma_bf16(qf[c].v, kf.v, s);
            }
            sfr[f] = s;
        }
        const int qrow = qb * 64 + wave * 16 + hs * 8;   // + v
#pragma unroll
        for (int f = 0; f < 2; ++f) {
            int key = key0 + f * 16 + l16;
            float bias = slope * (float)key;
#pragma unroll
            for (int v = 0; v < 8; ++v) {
                float sv = sfr[f][v] * scale + bias;
                sfr[f][v] = (key <= qrow + v) ? sv : -1.0e30f;
            }
        }
        float mnew[8], fac[8], rs[8];
#pragma unroll
        for (int v = 0; v < 8; ++v) {
            float mx = fmaxf(sfr[0][v], sfr[1][v]);
#pragma unroll
            for (int off = 8; off > 0; off >>= 1)
                mx = fmaxf(mx, __shfl_xor(mx, off, 32));
            mnew[v] = fmaxf(Mrow[v], mx);
            fac[v] = __expf(Mrow[v] - mnew[v]);
            rs[v] = 0.f;
        }
#pragma unroll
        for (int f = 0; f < 2; ++f)
#pragma unroll
            for (int v = 0; v < 8; ++v) {
                float p = __expf(sfr[f][v] - mnew[v]);
                sfr[f][v] = p;
                rs[v] += p;
            }
#pragma unroll
        for (int v = 0; v < 8; ++v) {
            float r = rs[v];
#pragma unroll
            for (int off = 8; off > 0; off >>= 1)
                r += __shfl_xor(r, off, 32);
            Lrow[v] = Lrow[v] * fac[v] + r;
            Mrow[v] = mnew[v];
        }
#pragma unroll
        for (int f = 0; f < 8; ++f)
#pragma unroll
            for (int v = 0; v < 8; ++v) O[f][v] *= fac[v];

        // C-layout -> A-fragment layout via per-wave LDS staging
        __bf16* pw = &Ps[wave][0];
#pragma unroll
        for (int f = 0; f < 2; ++f)
#pragma unroll
            for (int v = 0; v < 8; ++v)
                pw[(hs * 8 + v) * 40 + f * 16 + l16] = (__bf16)sfr[f][v];
        asm volatile("s_wait_dscnt 0" ::: "memory");
        BF16Frag pf;
        pf.q[0] = *(const uint4*)(pw + l16 * 40 + hs * 8);
        pf.q[1] = *(const uint4*)(pw + l16 * 40 + hs * 8 + 16);

#pragma unroll
        for (int f = 0; f < 8; ++f) {               // O += P x V (8 dim chunks)
            BF16Frag vf;
            int r = f * 16 + l16;
            vf.q[0] = *(const uint4*)(Vt + r * 40 + hs * 8);
            vf.q[1] = *(const uint4*)(Vt + r * 40 + hs * 8 + 16);
            O[f] = wmma_bf16(pf.v, vf.v, O[f]);
        }
        __syncthreads();
    }
    long long o0 = rowG + qb * 64 + wave * 16 + hs * 8;
#pragma unroll
    for (int f = 0; f < 8; ++f) {
        int col = h * 128 + f * 16 + l16;
#pragma unroll
        for (int v = 0; v < 8; ++v)
            attn[(o0 + v) * 5120 + col] = (__bf16)(O[f][v] / Lrow[v]);
    }
}

extern "C" void kernel_launch(void* const* d_in, const int* in_sizes, int n_in,
                              void* d_out, int out_size, void* d_ws, size_t ws_size,
                              hipStream_t stream) {
    const float* hidden = (const float*)d_in[0];
    // d_in[1] attention_mask is unused: ALiBi bias computed analytically on-chip
    const float* W_pack = (const float*)d_in[2];
    const float* o_proj = (const float*)d_in[3];
    float* out = (float*)d_out;

    __bf16* Xbf  = (__bf16*)d_ws;                    //  2048*5120
    __bf16* Wbf  = Xbf  + (long long)2048 * 5120;    // 15360*5120
    __bf16* Obf  = Wbf  + (long long)15360 * 5120;   //  5120*5120
    __bf16* QKV  = Obf  + (long long)5120 * 5120;    //  2048*15360
    __bf16* Attn = QKV  + (long long)2048 * 15360;   //  2048*5120

    auto cvt = [&](const float* s, __bf16* d, long long n) {
        long long n4 = n / 4;
        int blocks = (int)((n4 + 255) / 256);
        cvt_f32_bf16<<<blocks, 256, 0, stream>>>(s, d, n4);
    };
    cvt(hidden, Xbf, 2048LL * 5120);
    cvt(W_pack, Wbf, 15360LL * 5120);
    cvt(o_proj, Obf, 5120LL * 5120);

    gemm_xwt<false><<<dim3(15360 / 128, 2048 / 128), 256, 0, stream>>>(
        Xbf, Wbf, (void*)QKV, 2048, 15360, 5120);
    flash_attn<<<dim3(16, 40, 2), 128, 0, stream>>>(QKV, Attn);
    gemm_xwt<true><<<dim3(5120 / 128, 2048 / 128), 256, 0, stream>>>(
        Attn, Obf, (void*)out, 2048, 5120, 5120);
}
